// LongformerSelfAttention_33097017983002
// MI455X (gfx1250) — compile-verified, matrix operand order checked by reading, not run on hardware
//
#include <hip/hip_runtime.h>
#include <hip/hip_bf16.h>

// ---------------------------------------------------------------------------
// Longformer sliding-window self-attention, MI455X (gfx1250, wave32, WMMA).
//   B=2, S=4096, E=1024, H=16, D=64, W=256, C=S/W=16 chunks.
// Pipeline:
//   1) qkv_proj_kernel  x3 : bf16 WMMA GEMM  Y = X @ Wm^T + b  (Q scaled 1/8),
//                            output bf16 [B,H,S,D] into workspace.
//   2) sliding_attn_kernel : flash-style banded attention per (b,h,chunk),
//                            K/V sub-chunks staged in LDS (async global->LDS
//                            when the gfx1250 builtins exist), fp32 out [B,S,E].
// Workspace: 3 x 16 MB bf16 QKV = 48 MB.
// ---------------------------------------------------------------------------

typedef __bf16 bf16_t;
typedef __attribute__((ext_vector_type(16))) __bf16 v16bf;
typedef __attribute__((ext_vector_type(8)))  float  v8f;

#define WMMA_BF16(A, Bf, Cf) \
  __builtin_amdgcn_wmma_f32_16x16x32_bf16(false, (A), false, (Bf), (short)0, (Cf), false, false)

#define NEG_BIG (-3.0e38f)

// gfx1250 async global->LDS path (ASYNCcnt-tracked, bypasses VGPRs).
#if __has_builtin(__builtin_amdgcn_global_load_async_to_lds_b128) && \
    __has_builtin(__builtin_amdgcn_s_wait_asynccnt)
#define USE_ASYNC_LDS 1
typedef int v4i_t __attribute__((vector_size(16)));
typedef __attribute__((address_space(1))) v4i_t gv4i;   // global int4*
typedef __attribute__((address_space(3))) v4i_t lv4i;   // LDS    int4*
#else
#define USE_ASYNC_LDS 0
#endif

// ============================== QKV projection ==============================
// X  : [8192, 1024] fp32 (B*S rows)
// Wm : [1024, 1024] fp32, row-major [n][k]  (computes X @ Wm^T)
// out: [B, H, S, D] bf16;  n = h*64 + d, row m = b*4096 + s
// Block: 256 thr (8 waves). Wave computes 64(M) x 16(N); block tile 512 x 16.
__global__ __launch_bounds__(256) void qkv_proj_kernel(
    const float* __restrict__ X, const float* __restrict__ Wm,
    const float* __restrict__ bias, bf16_t* __restrict__ out, float scale) {
  const int lane = threadIdx.x & 31;
  const int wave = threadIdx.x >> 5;
  const int hf   = lane >> 4;       // half-wave select
  const int l16  = lane & 15;
  const int mbase = blockIdx.x * 512 + wave * 64;
  const int nbase = blockIdx.y * 16;
  const int k0    = hf * 8;         // per-lane K base inside a 32-wide slice

  v8f acc[4];
#pragma unroll
  for (int i = 0; i < 4; ++i) acc[i] = (v8f){0.f,0.f,0.f,0.f,0.f,0.f,0.f,0.f};

  const float* brow = Wm + (size_t)(nbase + l16) * 1024;
  for (int kk = 0; kk < 1024; kk += 32) {
    // B fragment: B[k][n] = Wm[n][k]  (lane = n, contiguous k reads)
    v16bf bfrag;
    {
      const float* src = brow + kk + k0;
      __builtin_prefetch(src + 64, 0, 1);   // global_prefetch_b8 (gfx1250)
#pragma unroll
      for (int i = 0; i < 8; ++i) {
        bfrag[i]     = (bf16_t)src[i];
        bfrag[8 + i] = (bf16_t)src[16 + i];
      }
    }
#pragma unroll
    for (int mt = 0; mt < 4; ++mt) {
      const float* asrc = X + (size_t)(mbase + mt * 16 + l16) * 1024 + kk + k0;
      v16bf afrag;
#pragma unroll
      for (int i = 0; i < 8; ++i) {
        afrag[i]     = (bf16_t)asrc[i];
        afrag[8 + i] = (bf16_t)asrc[16 + i];
      }
      acc[mt] = WMMA_BF16(afrag, bfrag, acc[mt]);
    }
  }

  const int   n  = nbase + l16;
  const float bn = bias[n];
  const int   h  = n >> 6, d = n & 63;
#pragma unroll
  for (int mt = 0; mt < 4; ++mt) {
#pragma unroll
    for (int r = 0; r < 8; ++r) {
      const int   gm = mbase + mt * 16 + r + hf * 8;   // C/D layout row
      const int   b  = gm >> 12, s = gm & 4095;
      const float v  = (acc[mt][r] + bn) * scale;
      out[(((size_t)b * 16 + h) * 4096 + s) * 64 + d] = (bf16_t)v;
    }
  }
}

// ========================= sliding-window attention =========================
// Grid: 512 blocks = (b,h,chunk). Block: 256 thr (8 waves).
// Wave owns 32 query rows (2 M-tiles). Flash softmax over 3 sub-chunks of 256
// keys; K staged [256][64], V staged transposed [64][256], P staged per-wave.
__global__ __launch_bounds__(256) void sliding_attn_kernel(
    const bf16_t* __restrict__ Q, const bf16_t* __restrict__ K,
    const bf16_t* __restrict__ V, float* __restrict__ out) {
  __shared__ bf16_t kbuf[256 * 64];     // 32 KB  K[key][d]
  __shared__ bf16_t vtb[64 * 256];      // 32 KB  V^T[d][key]
  __shared__ bf16_t pbuf[8][32 * 32];   // 16 KB  per-wave P staging

  const int tid  = threadIdx.x;
  const int lane = tid & 31, wave = tid >> 5;
  const int hf = lane >> 4, l16 = lane & 15;
  const int c = blockIdx.x & 15;
  const int h = (blockIdx.x >> 4) & 15;
  const int b = blockIdx.x >> 8;

  const size_t head = ((size_t)b * 16 + h) * 4096 * 64;
  const bf16_t* Qh = Q + head;
  const bf16_t* Kh = K + head;
  const bf16_t* Vh = V + head;

  const int k0 = hf * 8;

  // Q fragments (A-layout), rows qbase + l16, D=64 -> two K-slices of 32.
  v16bf qf[2][2];
#pragma unroll
  for (int mt = 0; mt < 2; ++mt) {
    const bf16_t* qrow = Qh + (size_t)(c * 256 + wave * 32 + mt * 16 + l16) * 64;
#pragma unroll
    for (int ks = 0; ks < 2; ++ks) {
      const int kb = ks * 32 + k0;
#pragma unroll
      for (int i = 0; i < 8; ++i) {
        qf[mt][ks][i]     = qrow[kb + i];
        qf[mt][ks][8 + i] = qrow[kb + 16 + i];
      }
    }
  }

  v8f oacc[2][4];
#pragma unroll
  for (int mt = 0; mt < 2; ++mt)
#pragma unroll
    for (int nt = 0; nt < 4; ++nt)
      oacc[mt][nt] = (v8f){0.f,0.f,0.f,0.f,0.f,0.f,0.f,0.f};
  float rmax[2][8], rsum[2][8];
#pragma unroll
  for (int mt = 0; mt < 2; ++mt)
#pragma unroll
    for (int r = 0; r < 8; ++r) { rmax[mt][r] = NEG_BIG; rsum[mt][r] = 0.f; }

  for (int sc = 0; sc < 3; ++sc) {
    __syncthreads();
    {  // stage one 256-key sub-chunk: each thread handles one key row
      const int key = tid;
      const int gk  = (c - 1 + sc) * 256 + key;
      if (gk >= 0 && gk < 4096) {
#if USE_ASYNC_LDS
        // K row: straight 128 B copy -> async global->LDS, no VGPR round trip.
        gv4i* gsrc = (gv4i*)(Kh + (size_t)gk * 64);
        lv4i* ldst = (lv4i*)&kbuf[key * 64];
#pragma unroll
        for (int i = 0; i < 8; ++i)
          __builtin_amdgcn_global_load_async_to_lds_b128(gsrc + i, ldst + i, 0, 0);
#else
        uint4*       kdst = reinterpret_cast<uint4*>(&kbuf[key * 64]);
        const uint4* ksrc = reinterpret_cast<const uint4*>(Kh + (size_t)gk * 64);
#pragma unroll
        for (int i = 0; i < 8; ++i) kdst[i] = ksrc[i];
#endif
        const bf16_t* vsrc = Vh + (size_t)gk * 64;
#pragma unroll
        for (int dd = 0; dd < 64; ++dd) vtb[dd * 256 + key] = vsrc[dd];
      } else {
        uint4*      kdst = reinterpret_cast<uint4*>(&kbuf[key * 64]);
        const uint4 z    = {0u, 0u, 0u, 0u};
#pragma unroll
        for (int i = 0; i < 8; ++i) kdst[i] = z;
#pragma unroll
        for (int dd = 0; dd < 64; ++dd) vtb[dd * 256 + key] = (bf16_t)0.f;
      }
    }
#if USE_ASYNC_LDS
    __builtin_amdgcn_s_wait_asynccnt(0);  // drain ASYNCcnt before the barrier
#endif
    __syncthreads();

    for (int pair = 0; pair < 8; ++pair) {  // 32 keys per step
      const int kb32 = pair * 32;

      // ---- scores: Q (16x64) . K^T -> two 16x16 tiles per M-tile ----
      v8f sacc[2][2];
#pragma unroll
      for (int mt = 0; mt < 2; ++mt)
#pragma unroll
        for (int ct = 0; ct < 2; ++ct)
          sacc[mt][ct] = (v8f){0.f,0.f,0.f,0.f,0.f,0.f,0.f,0.f};
#pragma unroll
      for (int ct = 0; ct < 2; ++ct) {
#pragma unroll
        for (int ks = 0; ks < 2; ++ks) {
          v16bf kf;  // B fragment: lane = key col, contiguous d reads
          const bf16_t* krow = &kbuf[(kb32 + ct * 16 + l16) * 64 + ks * 32 + k0];
#pragma unroll
          for (int i = 0; i < 8; ++i) { kf[i] = krow[i]; kf[8 + i] = krow[16 + i]; }
#pragma unroll
          for (int mt = 0; mt < 2; ++mt)
            sacc[mt][ct] = WMMA_BF16(qf[mt][ks], kf, sacc[mt][ct]);
        }
      }

      // ---- band mask + online softmax (stats per row, per 16-lane half) ----
#pragma unroll
      for (int mt = 0; mt < 2; ++mt) {
#pragma unroll
        for (int r = 0; r < 8; ++r) {
          const int qpos = c * 256 + wave * 32 + mt * 16 + r + hf * 8;
          float mx = NEG_BIG;
#pragma unroll
          for (int ct = 0; ct < 2; ++ct) {
            const int kpos = (c - 1 + sc) * 256 + kb32 + ct * 16 + l16;
            const bool ok = (kpos >= 0) && (kpos < 4096) &&
                            (kpos - qpos <= 256) && (qpos - kpos <= 256);
            const float s = ok ? sacc[mt][ct][r] : NEG_BIG;
            sacc[mt][ct][r] = s;
            mx = fmaxf(mx, s);
          }
          mx = fmaxf(mx, __shfl_xor(mx, 1, 32));
          mx = fmaxf(mx, __shfl_xor(mx, 2, 32));
          mx = fmaxf(mx, __shfl_xor(mx, 4, 32));
          mx = fmaxf(mx, __shfl_xor(mx, 8, 32));

          const float nm    = fmaxf(rmax[mt][r], mx);
          const float alpha = __expf(rmax[mt][r] - nm);
          rmax[mt][r] = nm;

          float psum = 0.f;
#pragma unroll
          for (int ct = 0; ct < 2; ++ct) {
            const float p = __expf(sacc[mt][ct][r] - nm);
            sacc[mt][ct][r] = p;
            psum += p;
          }
          psum += __shfl_xor(psum, 1, 32);
          psum += __shfl_xor(psum, 2, 32);
          psum += __shfl_xor(psum, 4, 32);
          psum += __shfl_xor(psum, 8, 32);
          rsum[mt][r] = rsum[mt][r] * alpha + psum;
#pragma unroll
          for (int nt = 0; nt < 4; ++nt) oacc[mt][nt][r] *= alpha;
        }
        // stage P (D-layout -> row-major) for A-layout reload
#pragma unroll
        for (int ct = 0; ct < 2; ++ct)
#pragma unroll
          for (int r = 0; r < 8; ++r)
            pbuf[wave][(mt * 16 + r + hf * 8) * 32 + ct * 16 + l16] =
                (bf16_t)sacc[mt][ct][r];
      }
      __syncthreads();  // pbuf WAR/RAW + keep next kbuf use ordered

      // ---- P (16x32) . V (32x64) accumulate ----
      v16bf pf[2];
#pragma unroll
      for (int mt = 0; mt < 2; ++mt) {
        const bf16_t* prow = &pbuf[wave][(mt * 16 + l16) * 32 + k0];
#pragma unroll
        for (int i = 0; i < 8; ++i) { pf[mt][i] = prow[i]; pf[mt][8 + i] = prow[16 + i]; }
      }
#pragma unroll
      for (int nt = 0; nt < 4; ++nt) {
        v16bf vf;  // B fragment: lane = d col, contiguous key reads from V^T
        const bf16_t* vrow = &vtb[(nt * 16 + l16) * 256 + kb32 + k0];
#pragma unroll
        for (int i = 0; i < 8; ++i) { vf[i] = vrow[i]; vf[8 + i] = vrow[16 + i]; }
#pragma unroll
        for (int mt = 0; mt < 2; ++mt)
          oacc[mt][nt] = WMMA_BF16(pf[mt], vf, oacc[mt][nt]);
      }
      __syncthreads();
    }
  }

  // ---- finalize: divide by row sums, store fp32 [B,S,H,D] ----
#pragma unroll
  for (int mt = 0; mt < 2; ++mt) {
#pragma unroll
    for (int r = 0; r < 8; ++r) {
      const float inv = 1.f / rsum[mt][r];  // own key always valid -> > 0
      const int   s   = c * 256 + wave * 32 + mt * 16 + r + hf * 8;
#pragma unroll
      for (int nt = 0; nt < 4; ++nt) {
        const int d = nt * 16 + l16;
        out[(((size_t)b * 4096 + s) * 16 + h) * 64 + d] = oacc[mt][nt][r] * inv;
      }
    }
  }
}

// ================================= launch ==================================
extern "C" void kernel_launch(void* const* d_in, const int* in_sizes, int n_in,
                              void* d_out, int out_size, void* d_ws, size_t ws_size,
                              hipStream_t stream) {
  (void)in_sizes; (void)n_in; (void)out_size; (void)ws_size;
  const float* X  = (const float*)d_in[0];
  const float* Wq = (const float*)d_in[1];
  const float* bq = (const float*)d_in[2];
  const float* Wk = (const float*)d_in[3];
  const float* bk = (const float*)d_in[4];
  const float* Wv = (const float*)d_in[5];
  const float* bv = (const float*)d_in[6];

  bf16_t* qb = (bf16_t*)d_ws;           // [2,16,4096,64] bf16 = 16 MB
  bf16_t* kb = qb + (size_t)8388608;
  bf16_t* vb = kb + (size_t)8388608;

  const dim3 ggrid(16, 64, 1);          // M 8192/512, N 1024/16
  qkv_proj_kernel<<<ggrid, 256, 0, stream>>>(X, Wq, bq, qb, 0.125f);
  qkv_proj_kernel<<<ggrid, 256, 0, stream>>>(X, Wk, bk, kb, 1.0f);
  qkv_proj_kernel<<<ggrid, 256, 0, stream>>>(X, Wv, bv, vb, 1.0f);

  sliding_attn_kernel<<<512, 256, 0, stream>>>(qb, kb, vb, (float*)d_out);
}
